// sigmod_61418032333301
// MI455X (gfx1250) — compile-verified, matrix operand order checked by reading, not run on hardware
//
#include <hip/hip_runtime.h>
#include <cstdint>

// GRU forward for MI455X (gfx1250, wave32, WMMA).
// Strategy: fuse embedding-gather + input GEMM into each recurrent step,
// accumulate r/z gates' input-path and hidden-path GEMMs into the SAME
// v_wmma_f32_16x16x32_bf16 accumulators, keep weights as pre-transposed bf16
// (L2-resident), ping-pong f32 hidden state in workspace across 100 step
// kernel launches.

#define EMB    256
#define HID    512
#define HID3   1536
#define BATCH  4096
#define SEQT   100
#define VOCABN 50000

#define BM 32                 // rows of h_new per block
#define BN 64                 // cols of h_new per block
#define XE_STR (EMB + 8)      // LDS row stride (bf16 elems), 16B-aligned rows
#define H_STR  (HID + 8)

typedef __attribute__((ext_vector_type(16))) __bf16 v16bf;
typedef __attribute__((ext_vector_type(8)))  float  v8f;

union BfFrag { uint4 q[2]; v16bf v; };

__device__ __forceinline__ unsigned f32_to_bf16_bits(float f) {
  unsigned u = __float_as_uint(f);
  return (u + 0x7FFFu + ((u >> 16) & 1u)) >> 16;   // round-to-nearest-even
}
__device__ __forceinline__ unsigned pack2(float lo, float hi) {
  return f32_to_bf16_bits(lo) | (f32_to_bf16_bits(hi) << 16);
}
__device__ __forceinline__ float fast_sigmoid(float x) {
  return 1.0f / (1.0f + __expf(-x));
}
__device__ __forceinline__ float fast_tanh(float x) {
  return 2.0f / (1.0f + __expf(-2.0f * x)) - 1.0f;
}
__device__ __forceinline__ v8f zero8() {
  v8f z;
#pragma unroll
  for (int i = 0; i < 8; ++i) z[i] = 0.0f;
  return z;
}

// A-operand fragment (16x32 bf16, M x K) from LDS, per ISA 7.12.2 layout:
// lanes 0-15: row M=lane, K = {k0..k0+7, k0+16..k0+23}
// lanes 16-31: row M=lane-16, K = {k0+8..k0+15, k0+24..k0+31}
__device__ __forceinline__ v16bf load_frag_a_lds(const unsigned short* base,
                                                 int stride, int lane) {
  const unsigned short* p = base + (lane & 15) * stride + ((lane >> 4) << 3);
  BfFrag f;
  f.q[0] = *(const uint4*)(p);
  f.q[1] = *(const uint4*)(p + 16);
  return f.v;
}

// B-operand fragment (32x16 bf16, K x N) from pre-transposed weights Wt (N x K
// row-major): lane n holds column N=col+(lane&15), same K split as A.
__device__ __forceinline__ v16bf load_frag_b_glb(const unsigned short* __restrict__ Wt,
                                                 int col, int K, int k0, int lane) {
  const unsigned short* p =
      Wt + (size_t)(col + (lane & 15)) * K + k0 + ((lane >> 4) << 3);
  BfFrag f;
  f.q[0] = *(const uint4*)(p);
  f.q[1] = *(const uint4*)(p + 16);
  return f.v;
}

// W (K x 1536 f32, row-major) -> Wt (1536 x K bf16, row-major)
__global__ void wt_to_bf16_t(const float* __restrict__ W,
                             unsigned short* __restrict__ Wt, int K) {
  int idx = blockIdx.x * blockDim.x + threadIdx.x;
  if (idx >= K * HID3) return;
  int n = idx / K;
  int e = idx - n * K;
  Wt[idx] = (unsigned short)f32_to_bf16_bits(W[(size_t)e * HID3 + n]);
}

__global__ __launch_bounds__(128)
void gru_step(const long long* __restrict__ x, const float* __restrict__ emb,
              const unsigned short* __restrict__ WiT,
              const unsigned short* __restrict__ WhT,
              const float* __restrict__ b_in, const float* __restrict__ b_h,
              const float* __restrict__ h_in, float* __restrict__ h_out, int t) {
  extern __shared__ unsigned short smem[];
  unsigned short* sh_xe = smem;                 // BM x XE_STR bf16
  unsigned short* sh_h  = smem + BM * XE_STR;   // BM x H_STR bf16
  __shared__ int sh_tok[BM];

  const int tid = threadIdx.x;
  const int m0 = blockIdx.x * BM;
  const int n0 = blockIdx.y * BN;

  if (tid < BM) {
    long long tk = x[(size_t)(m0 + tid) * SEQT + t];
    if (tk < 0) tk = 0;
    if (tk >= VOCABN) tk = VOCABN - 1;
    sh_tok[tid] = (int)tk;
  }
  __syncthreads();

  // Stage gathered embedding rows, f32 -> bf16 (table is L2-resident: 51 MB < 192 MB).
  for (int i = tid; i < BM * (EMB / 4); i += 128) {
    int r = i >> 6;              // / (EMB/4)
    int c4 = i & 63;
    float4 v = ((const float4*)emb)[(size_t)sh_tok[r] * (EMB / 4) + c4];
    uint2 p; p.x = pack2(v.x, v.y); p.y = pack2(v.z, v.w);
    *(uint2*)(sh_xe + r * XE_STR + c4 * 4) = p;
  }
  // Stage hidden-state rows, f32 -> bf16.
  for (int i = tid; i < BM * (HID / 4); i += 128) {
    int r = i >> 7;              // / (HID/4)
    int c4 = i & 127;
    float4 v = ((const float4*)h_in)[(size_t)(m0 + r) * (HID / 4) + c4];
    uint2 p; p.x = pack2(v.x, v.y); p.y = pack2(v.z, v.w);
    *(uint2*)(sh_h + r * H_STR + c4 * 4) = p;
  }
  __syncthreads();

  const int lane = tid & 31;
  const int wave = tid >> 5;
  const int wm = wave & 1;       // 2 M-tiles of 16
  const int wn = wave >> 1;      // 2 N-tiles of 32

  // r/z: input-path and hidden-path share accumulators (sigmoid(ti+th)).
  // n: must stay split because of n = tanh(n_i + r * n_h).
  v8f accR[2], accZ[2], accNi[2], accNh[2];
#pragma unroll
  for (int s = 0; s < 2; ++s) {
    accR[s] = zero8(); accZ[s] = zero8(); accNi[s] = zero8(); accNh[s] = zero8();
  }

  // ---- input path: xe(BMx256) @ W_in(256x1536), K=256 ----
  const unsigned short* aIn = sh_xe + wm * 16 * XE_STR;
  for (int k0 = 0; k0 < EMB; k0 += 32) {
    v16bf a = load_frag_a_lds(aIn + k0, XE_STR, lane);
#pragma unroll
    for (int g = 0; g < 3; ++g) {
#pragma unroll
      for (int s = 0; s < 2; ++s) {
        int col = g * HID + n0 + wn * 32 + s * 16;
        v16bf b = load_frag_b_glb(WiT, col, EMB, k0, lane);
        v8f& acc = (g == 0) ? accR[s] : (g == 1) ? accZ[s] : accNi[s];
        acc = __builtin_amdgcn_wmma_f32_16x16x32_bf16(
            false, a, false, b, (short)0, acc, false, false);
      }
    }
  }

  // ---- hidden path: h(BMx512) @ W_h(512x1536), K=512 ----
  const unsigned short* aH = sh_h + wm * 16 * H_STR;
  for (int k0 = 0; k0 < HID; k0 += 32) {
    v16bf a = load_frag_a_lds(aH + k0, H_STR, lane);
#pragma unroll
    for (int g = 0; g < 3; ++g) {
#pragma unroll
      for (int s = 0; s < 2; ++s) {
        int col = g * HID + n0 + wn * 32 + s * 16;
        v16bf b = load_frag_b_glb(WhT, col, HID, k0, lane);
        v8f& acc = (g == 0) ? accR[s] : (g == 1) ? accZ[s] : accNh[s];
        acc = __builtin_amdgcn_wmma_f32_16x16x32_bf16(
            false, a, false, b, (short)0, acc, false, false);
      }
    }
  }

  // ---- gate math + h update. C/D layout (ISA 7.12.2):
  // lane 0-15: N = lane, M = vgpr e (0..7); lane 16-31: N = lane-16, M = 8+e.
#pragma unroll
  for (int s = 0; s < 2; ++s) {
    int col = n0 + wn * 32 + s * 16 + (lane & 15);      // 0..511
    float bR  = b_in[col]           + b_h[col];
    float bZ  = b_in[col + HID]     + b_h[col + HID];
    float bNi = b_in[col + 2 * HID];
    float bNh = b_h [col + 2 * HID];
    int mbase = m0 + wm * 16 + ((lane >> 4) << 3);
#pragma unroll
    for (int e = 0; e < 8; ++e) {
      int m = mbase + e;
      float r  = fast_sigmoid(accR[s][e] + bR);
      float zg = fast_sigmoid(accZ[s][e] + bZ);
      float nn = fast_tanh(accNi[s][e] + bNi + r * (accNh[s][e] + bNh));
      float ho = h_in[(size_t)m * HID + col];
      h_out[(size_t)m * HID + col] = (1.0f - zg) * nn + zg * ho;
    }
  }
}

// out = tanh(h @ W1 + b1) @ W2 + b2   (tiny: one row per thread)
__global__ __launch_bounds__(128)
void gru_head(const float* __restrict__ h, const float* __restrict__ W1,
              const float* __restrict__ b1, const float* __restrict__ W2,
              const float* __restrict__ b2, float* __restrict__ out) {
  int row = blockIdx.x * blockDim.x + threadIdx.x;
  if (row >= BATCH) return;
  float acc[16];
#pragma unroll
  for (int j = 0; j < 16; ++j) acc[j] = b1[j];
  for (int k = 0; k < HID; ++k) {
    float hv = h[(size_t)row * HID + k];
#pragma unroll
    for (int j = 0; j < 16; ++j) acc[j] += hv * W1[k * 16 + j];
  }
  float o0 = b2[0], o1 = b2[1];
#pragma unroll
  for (int j = 0; j < 16; ++j) {
    float u = fast_tanh(acc[j]);
    o0 += u * W2[j * 2 + 0];
    o1 += u * W2[j * 2 + 1];
  }
  out[(size_t)row * 2 + 0] = o0;
  out[(size_t)row * 2 + 1] = o1;
}

extern "C" void kernel_launch(void* const* d_in, const int* in_sizes, int n_in,
                              void* d_out, int out_size, void* d_ws, size_t ws_size,
                              hipStream_t stream) {
  (void)in_sizes; (void)n_in; (void)out_size; (void)ws_size;

  const long long* x    = (const long long*)d_in[0];   // int64 per reference
  const float*     emb  = (const float*)d_in[1];
  const float*     W_in = (const float*)d_in[2];
  const float*     b_in = (const float*)d_in[3];
  const float*     W_h  = (const float*)d_in[4];
  const float*     b_h  = (const float*)d_in[5];
  const float*     W1   = (const float*)d_in[6];
  const float*     b1   = (const float*)d_in[7];
  const float*     W2   = (const float*)d_in[8];
  const float*     b2   = (const float*)d_in[9];
  float*           out  = (float*)d_out;

  // Workspace layout (~18.5 MB used):
  char* ws = (char*)d_ws;
  const size_t hBytes = (size_t)BATCH * HID * sizeof(float);    // 8 MB
  float* hA = (float*)ws;
  float* hB = (float*)(ws + hBytes);
  unsigned short* WiT = (unsigned short*)(ws + 2 * hBytes);                    // 1536x256 bf16
  unsigned short* WhT = (unsigned short*)(ws + 2 * hBytes +
                                          (size_t)HID3 * EMB * sizeof(unsigned short));

  // h0 = 0
  hipMemsetAsync(hA, 0, hBytes, stream);

  // Pre-transpose + convert weights to bf16 (N x K), done once per launch.
  wt_to_bf16_t<<<(HID3 * EMB + 255) / 256, 256, 0, stream>>>(W_in, WiT, EMB);
  wt_to_bf16_t<<<(HID3 * HID + 255) / 256, 256, 0, stream>>>(W_h,  WhT, HID);

  const size_t ldsBytes = (size_t)BM * (XE_STR + H_STR) * sizeof(unsigned short); // ~50 KB
  dim3 grid(BATCH / BM, HID / BN);   // (128, 8)
  dim3 block(128);

  const float* hin = hA;
  float* hout = hB;
  for (int t = 0; t < SEQT; ++t) {
    gru_step<<<grid, block, ldsBytes, stream>>>(x, emb, WiT, WhT, b_in, b_h,
                                                hin, hout, t);
    const float* tmp = hout;
    hout = (float*)hin;
    hin = tmp;
  }
  // After an even number (100) of swaps, the final hidden state is in `hin` (== hA).
  gru_head<<<BATCH / 128, 128, 0, stream>>>(hin, W1, b1, W2, b2, out);
}
